// GraphSeqDiscriminator_77799037599898
// MI455X (gfx1250) — matrix operand order repaired and checked by reading, not compiled, not run on hardware
//
#include <hip/hip_runtime.h>
#include <hip/hip_bf16.h>

// ---------------------------------------------------------------- constants
#define N_NODES 50000
#define E_EDGES 800000
#define FDIM    128
#define LDIM    64
#define MTILE   64        // rows per block
#define SUBT    4         // 16-row M-subtiles per wave

typedef __attribute__((ext_vector_type(16))) __bf16 v16bf;
typedef __attribute__((ext_vector_type(8)))  __bf16 bf16x8;
typedef __attribute__((ext_vector_type(4)))  __bf16 bf16x4;
typedef __attribute__((ext_vector_type(8)))  float  v8f;

// Packed bf16 weight layout (element offsets):
//  WZp  @ 0       : 512x128  (Wxz0|Wxz1|Whz0|Whz1)
//  WRp  @ 65536   : 512x128
//  WHxp @ 131072  : 256x128  (Wxh0|Wxh1)
//  WHhp @ 163840  : 256x128  (Whh0|Whh1)
//  WLp  @ 196608  : 128x64   (Wlin)
#define PACKED_TOTAL 204800

#define ASTRIDE  520      // 512 K + 8 pad (bf16), 16B-aligned rows
#define A2STRIDE 264      // 256 K + 8 pad
#define H0STRIDE 136      // 128 K + 8 pad

__device__ __forceinline__ float sigmoidf_(float v) {
    return 1.f / (1.f + __expf(-v));
}

// A-fragment: lane holds row M=lane%16; 16 bf16 = two runs of 8 along K.
__device__ __forceinline__ v16bf load_a_frag(const __bf16* p) {
    bf16x8 lo = *reinterpret_cast<const bf16x8*>(p);
    bf16x8 hi = *reinterpret_cast<const bf16x8*>(p + 16);
    v16bf r;
#pragma unroll
    for (int i = 0; i < 8; ++i) { r[i] = lo[i]; r[8 + i] = hi[i]; }
    return r;
}

// B-fragment: pre-swizzled, 16 contiguous bf16 per lane.
__device__ __forceinline__ v16bf load_b_frag(const __bf16* p) {
    bf16x8 lo = *reinterpret_cast<const bf16x8*>(p);
    bf16x8 hi = *reinterpret_cast<const bf16x8*>(p + 8);
    v16bf r;
#pragma unroll
    for (int i = 0; i < 8; ++i) { r[i] = lo[i]; r[8 + i] = hi[i]; }
    return r;
}

#define WMMA_BF16(a, b, c) \
    __builtin_amdgcn_wmma_f32_16x16x32_bf16(false, (a), false, (b), (short)0, (c), false, false)

// ------------------------------------------------------------ weight packing
__global__ void pack_weights_kernel(
    const float* __restrict__ Wxz, const float* __restrict__ Whz,
    const float* __restrict__ Wxr, const float* __restrict__ Whr,
    const float* __restrict__ Wxh, const float* __restrict__ Whh,
    const float* __restrict__ Wlin,
    const float* __restrict__ bxz, const float* __restrict__ bhz,
    const float* __restrict__ bxr, const float* __restrict__ bhr,
    const float* __restrict__ bxh, const float* __restrict__ bhh,
    const float* __restrict__ blin,
    __bf16* __restrict__ packed, float* __restrict__ bias)
{
    int t = blockIdx.x * blockDim.x + threadIdx.x;
    if (t < 128) {
        bias[t]       = bxz[t] + bhz[t];
        bias[128 + t] = bxr[t] + bhr[t];
        bias[256 + t] = bxh[t] + bhh[t];
        if (t < 64) bias[384 + t] = blin[t];
    }
    if (t >= PACKED_TOTAL) return;

    int off, Nc;
    const float* S0;
    const float* S1 = nullptr;
    if (t < 65536)       { off = t;          Nc = 128; S0 = Wxz; S1 = Whz; }
    else if (t < 131072) { off = t - 65536;  Nc = 128; S0 = Wxr; S1 = Whr; }
    else if (t < 163840) { off = t - 131072; Nc = 128; S0 = Wxh; }
    else if (t < 196608) { off = t - 163840; Nc = 128; S0 = Whh; }
    else                 { off = t - 196608; Nc = 64;  S0 = Wlin; }

    int frag = off >> 9, rem = off & 511;
    int lane = rem >> 4, i = rem & 15;
    int nnt = Nc >> 4;
    int kt = frag / nnt, nt = frag % nnt;
    // B-matrix 32x16 layout: lane holds col N=lane%16, K = 8*(lane/16)+i (i<8), 16+8*(lane/16)+(i-8) (i>=8)
    int k = kt * 32 + ((lane >> 4) << 3) + ((i < 8) ? i : (i + 8));
    int n = nt * 16 + (lane & 15);
    const float* S = (k < 256) ? S0 : S1;
    int kk = (k < 256) ? k : (k - 256);
    packed[t] = (__bf16)S[kk * Nc + n];
}

// ------------------------------------------------------------- graph kernels
__global__ void deg_kernel(const long long* __restrict__ ei,
                           const float* __restrict__ ew,
                           float* __restrict__ deg)
{
    int e = blockIdx.x * blockDim.x + threadIdx.x;
    if (e < E_EDGES) atomicAdd(&deg[(int)ei[e]], ew[e]);
}

__global__ void norm_kernel(const long long* __restrict__ ei,
                            const float* __restrict__ ew,
                            const float* __restrict__ deg,
                            float* __restrict__ w)
{
    int e = blockIdx.x * blockDim.x + threadIdx.x;
    if (e >= E_EDGES) return;
    int s = (int)ei[e], d = (int)ei[E_EDGES + e];
    float ds = deg[s], dd = deg[d];
    float rs = ds > 0.f ? rsqrtf(ds) : 0.f;
    float rd = dd > 0.f ? rsqrtf(dd) : 0.f;
    w[e] = -rs * ew[e] * rd;
}

// Fused scatter: Lx[dst] += w*x[src], Lh[dst] += w*h[src]. One wave per edge.
__global__ void scatter2_kernel(const long long* __restrict__ ei,
                                const float* __restrict__ w,
                                const float* __restrict__ x,
                                const float* __restrict__ h,
                                float* __restrict__ Lx,
                                float* __restrict__ Lh)
{
    int gid = blockIdx.x * blockDim.x + threadIdx.x;
    int e = gid >> 5, lane = gid & 31;
    if (e >= E_EDGES) return;
    float we = w[e];
    int s = (int)ei[e], d = (int)ei[E_EDGES + e];
    float4 xv = reinterpret_cast<const float4*>(x + (size_t)s * FDIM)[lane];
    float4 hv = reinterpret_cast<const float4*>(h + (size_t)s * FDIM)[lane];
    float* px = Lx + (size_t)d * FDIM + lane * 4;
    float* ph = Lh + (size_t)d * FDIM + lane * 4;
    atomicAdd(px + 0, we * xv.x); atomicAdd(px + 1, we * xv.y);
    atomicAdd(px + 2, we * xv.z); atomicAdd(px + 3, we * xv.w);
    atomicAdd(ph + 0, we * hv.x); atomicAdd(ph + 1, we * hv.y);
    atomicAdd(ph + 2, we * hv.z); atomicAdd(ph + 3, we * hv.w);
}

__global__ void scatter1_kernel(const long long* __restrict__ ei,
                                const float* __restrict__ w,
                                const float* __restrict__ hr,
                                float* __restrict__ Lhr)
{
    int gid = blockIdx.x * blockDim.x + threadIdx.x;
    int e = gid >> 5, lane = gid & 31;
    if (e >= E_EDGES) return;
    float we = w[e];
    int s = (int)ei[e], d = (int)ei[E_EDGES + e];
    float4 v = reinterpret_cast<const float4*>(hr + (size_t)s * FDIM)[lane];
    float* p = Lhr + (size_t)d * FDIM + lane * 4;
    atomicAdd(p + 0, we * v.x); atomicAdd(p + 1, we * v.y);
    atomicAdd(p + 2, we * v.z); atomicAdd(p + 3, we * v.w);
}

// -------------------------------------------------------------- GEMM stage A
// A = [x | Lx | h | Lh] (K=512), 64 rows/block, 4 M-subtiles per wave.
// Outputs: Z=sig(.), R=sig(.), preHx, hr=h*R (hr aliases Lh rows of this block).
__global__ __launch_bounds__(256) void gemm_gates_kernel(
    const float* __restrict__ x, const float* __restrict__ h,
    const float* __restrict__ Lx, const float* __restrict__ Lh,
    const __bf16* __restrict__ packed, const float* __restrict__ bias,
    float* __restrict__ Zbuf, float* __restrict__ preHx,
    float* __restrict__ hrbuf)
{
    __shared__ __align__(16) __bf16 As[MTILE * ASTRIDE];
    int Mbase = blockIdx.x * MTILE;

    // Stage [x|Lx|h|Lh] rows -> bf16 LDS, float4-vectorized.
    for (int j = threadIdx.x; j < 4 * MTILE * 32; j += 256) {
        int seg = j >> 11;          // 2048 float4 per 64x128 segment
        int rj = j & 2047;
        int row = rj >> 5, c4 = rj & 31;
        int node = Mbase + row;
        float4 v = make_float4(0.f, 0.f, 0.f, 0.f);
        if (node < N_NODES) {
            const float* s = (seg == 0) ? x : (seg == 1) ? Lx : (seg == 2) ? h : Lh;
            v = reinterpret_cast<const float4*>(s + (size_t)node * FDIM)[c4];
        }
        bf16x4 b; b[0] = (__bf16)v.x; b[1] = (__bf16)v.y;
                  b[2] = (__bf16)v.z; b[3] = (__bf16)v.w;
        *reinterpret_cast<bf16x4*>(&As[row * ASTRIDE + seg * 128 + c4 * 4]) = b;
    }
    __syncthreads();

    int wave = threadIdx.x >> 5, lane = threadIdx.x & 31;
    int rowA = lane & 15, hsel = lane >> 4;
    const __bf16* WZ  = packed;
    const __bf16* WR  = packed + 65536;
    const __bf16* WHx = packed + 131072;
    int nt = wave;

    v8f cZ[SUBT], cR[SUBT], cH[SUBT];
#pragma unroll
    for (int s = 0; s < SUBT; ++s) { cZ[s] = (v8f){}; cR[s] = (v8f){}; cH[s] = (v8f){}; }

#pragma unroll
    for (int kt = 0; kt < 16; ++kt) {
        v16bf bz = load_b_frag(WZ + (((kt * 8 + nt) << 9) + (lane << 4)));
        v16bf br = load_b_frag(WR + (((kt * 8 + nt) << 9) + (lane << 4)));
#pragma unroll
        for (int s = 0; s < SUBT; ++s) {
            v16bf a = load_a_frag(&As[(s * 16 + rowA) * ASTRIDE + hsel * 8 + kt * 32]);
            cZ[s] = WMMA_BF16(a, bz, cZ[s]);
            cR[s] = WMMA_BF16(a, br, cR[s]);
            if (kt < 8) {  // preHx consumes K=[0,256): x,Lx only
                v16bf bh = load_b_frag(WHx + (((kt * 8 + nt) << 9) + (lane << 4)));
                cH[s] = WMMA_BF16(a, bh, cH[s]);
            }
        }
    }

    int ncol = nt * 16 + (lane & 15);
    float bZv = bias[ncol], bRv = bias[128 + ncol];
#pragma unroll
    for (int s = 0; s < SUBT; ++s) {
#pragma unroll
        for (int r = 0; r < 8; ++r) {
            int node = Mbase + s * 16 + hsel * 8 + r;
            if (node < N_NODES) {
                size_t o = (size_t)node * FDIM + ncol;
                float zg = sigmoidf_(cZ[s][r] + bZv);
                float rg = sigmoidf_(cR[s][r] + bRv);
                Zbuf[o]  = zg;
                preHx[o] = cH[s][r];
                hrbuf[o] = h[o] * rg;
            }
        }
    }
}

// -------------------------------------------------------------- GEMM stage B
// Ht = tanh(preHx + [hr|Lhr]@Whh + bH); h0 = Z*h+(1-Z)*Ht (-> d_out);
// z = relu(relu(h0)@Wlin + blin); colsum += column-sums of z.
__global__ __launch_bounds__(256) void gemm_out_kernel(
    const float* __restrict__ h, const float* __restrict__ hr,
    const float* __restrict__ Lhr, const float* __restrict__ Zbuf,
    const float* __restrict__ preHx,
    const __bf16* __restrict__ packed, const float* __restrict__ bias,
    float* __restrict__ h0out, float* __restrict__ colsum)
{
    __shared__ __align__(16) __bf16 As[MTILE * A2STRIDE];
    __shared__ __align__(16) __bf16 H0[MTILE * H0STRIDE];
    int Mbase = blockIdx.x * MTILE;

    for (int j = threadIdx.x; j < 2 * MTILE * 32; j += 256) {
        int seg = j >> 11;
        int rj = j & 2047;
        int row = rj >> 5, c4 = rj & 31;
        int node = Mbase + row;
        float4 v = make_float4(0.f, 0.f, 0.f, 0.f);
        if (node < N_NODES) {
            const float* s = (seg == 0) ? hr : Lhr;
            v = reinterpret_cast<const float4*>(s + (size_t)node * FDIM)[c4];
        }
        bf16x4 b; b[0] = (__bf16)v.x; b[1] = (__bf16)v.y;
                  b[2] = (__bf16)v.z; b[3] = (__bf16)v.w;
        *reinterpret_cast<bf16x4*>(&As[row * A2STRIDE + seg * 128 + c4 * 4]) = b;
    }
    __syncthreads();

    int wave = threadIdx.x >> 5, lane = threadIdx.x & 31;
    int rowA = lane & 15, hsel = lane >> 4;
    const __bf16* WHh = packed + 163840;
    const __bf16* WL  = packed + 196608;
    int nt = wave;

    v8f c[SUBT];
#pragma unroll
    for (int s = 0; s < SUBT; ++s) c[s] = (v8f){};

#pragma unroll
    for (int kt = 0; kt < 8; ++kt) {
        v16bf b = load_b_frag(WHh + (((kt * 8 + nt) << 9) + (lane << 4)));
#pragma unroll
        for (int s = 0; s < SUBT; ++s) {
            v16bf a = load_a_frag(&As[(s * 16 + rowA) * A2STRIDE + hsel * 8 + kt * 32]);
            c[s] = WMMA_BF16(a, b, c[s]);
        }
    }

    int ncol = nt * 16 + (lane & 15);
    float bHv = bias[256 + ncol];
#pragma unroll
    for (int s = 0; s < SUBT; ++s) {
#pragma unroll
        for (int r = 0; r < 8; ++r) {
            int m = s * 16 + hsel * 8 + r;
            int node = Mbase + m;
            float relv = 0.f;
            if (node < N_NODES) {
                size_t o = (size_t)node * FDIM + ncol;
                float Ht  = tanhf(preHx[o] + c[s][r] + bHv);
                float zg  = Zbuf[o];
                float h0v = zg * h[o] + (1.f - zg) * Ht;
                h0out[o] = h0v;
                relv = h0v > 0.f ? h0v : 0.f;
            }
            H0[m * H0STRIDE + ncol] = (__bf16)relv;
        }
    }
    __syncthreads();

    if (wave < 4) {  // 4 waves cover 64-wide Wlin output; wave-uniform branch
        int nt2 = wave;
        v8f cz[SUBT];
#pragma unroll
        for (int s = 0; s < SUBT; ++s) cz[s] = (v8f){};
#pragma unroll
        for (int kt = 0; kt < 4; ++kt) {
            v16bf b = load_b_frag(WL + (((kt * 4 + nt2) << 9) + (lane << 4)));
#pragma unroll
            for (int s = 0; s < SUBT; ++s) {
                v16bf a = load_a_frag(&H0[(s * 16 + rowA) * H0STRIDE + hsel * 8 + kt * 32]);
                cz[s] = WMMA_BF16(a, b, cz[s]);
            }
        }
        int nc2 = nt2 * 16 + (lane & 15);
        float bl = bias[384 + nc2];
        float acc = 0.f;
#pragma unroll
        for (int s = 0; s < SUBT; ++s) {
#pragma unroll
            for (int r = 0; r < 8; ++r) {
                int node = Mbase + s * 16 + hsel * 8 + r;
                if (node < N_NODES) {
                    float zv = cz[s][r] + bl;
                    acc += zv > 0.f ? zv : 0.f;
                }
            }
        }
        acc += __shfl_xor(acc, 16, 32);      // fold the two half-waves
        if (lane < 16) atomicAdd(&colsum[nc2], acc);
    }
}

__global__ void finalize_kernel(const float* __restrict__ colsum,
                                float* __restrict__ out)
{
    int t = blockIdx.x * blockDim.x + threadIdx.x;
    if (t < LDIM) out[t] = sigmoidf_(colsum[t] * (1.f / (float)N_NODES));
}

// ------------------------------------------------------------------ launcher
extern "C" void kernel_launch(void* const* d_in, const int* in_sizes, int n_in,
                              void* d_out, int out_size, void* d_ws, size_t ws_size,
                              hipStream_t stream)
{
    const float*     x    = (const float*)d_in[0];
    const long long* ei   = (const long long*)d_in[1];
    const float*     ew   = (const float*)d_in[2];
    const float*     h    = (const float*)d_in[3];
    const float* Wxz = (const float*)d_in[4];  const float* bxz = (const float*)d_in[5];
    const float* Whz = (const float*)d_in[6];  const float* bhz = (const float*)d_in[7];
    const float* Wxr = (const float*)d_in[8];  const float* bxr = (const float*)d_in[9];
    const float* Whr = (const float*)d_in[10]; const float* bhr = (const float*)d_in[11];
    const float* Wxh = (const float*)d_in[12]; const float* bxh = (const float*)d_in[13];
    const float* Whh = (const float*)d_in[14]; const float* bhh = (const float*)d_in[15];
    const float* Wlin = (const float*)d_in[16]; const float* blin = (const float*)d_in[17];
    (void)in_sizes; (void)n_in; (void)out_size; (void)ws_size;

    // Workspace layout (float offsets)
    float* ws = (float*)d_ws;
    const size_t NF = (size_t)N_NODES * FDIM;                     // 6,400,000
    float* deg    = ws;                                           // N
    float* colsum = ws + N_NODES;                                 // 64
    float* bias   = ws + N_NODES + 64;                            // 448
    float* Lx     = ws + N_NODES + 512;                           // N*F
    float* Lh     = Lx + NF;                                      // N*F
    float* Zbuf   = Lh + NF;                                      // N*F
    float* preHx  = Zbuf + NF;                                    // N*F
    float* wnorm  = preHx + NF;                                   // E
    __bf16* packed = (__bf16*)(wnorm + E_EDGES);                  // 204800 bf16
    float* hrbuf  = Lh;   // reuse: Lh dead after gemm_gates (in-place per row)
    float* Lhr    = Lx;   // reuse: Lx dead after gemm_gates

    float* out_head = (float*)d_out;        // 64 floats
    float* out_h0   = out_head + LDIM;      // N*F floats

    // 1) zero deg + colsum + bias + Lx + Lh in one contiguous memset
    hipMemsetAsync(ws, 0, (N_NODES + 512 + 2 * NF) * sizeof(float), stream);

    // 2) pack weights (bf16, WMMA B-fragment order) + fold biases
    pack_weights_kernel<<<(PACKED_TOTAL + 255) / 256, 256, 0, stream>>>(
        Wxz, Whz, Wxr, Whr, Wxh, Whh, Wlin,
        bxz, bhz, bxr, bhr, bxh, bhh, blin, packed, bias);

    // 3) degrees, 4) edge norms
    deg_kernel<<<(E_EDGES + 255) / 256, 256, 0, stream>>>(ei, ew, deg);
    norm_kernel<<<(E_EDGES + 255) / 256, 256, 0, stream>>>(ei, ew, deg, wnorm);

    // 5) Lx = L_hat x, Lh = L_hat h (fused, wave per edge)
    scatter2_kernel<<<(E_EDGES * 32) / 256, 256, 0, stream>>>(ei, wnorm, x, h, Lx, Lh);

    // 6) gates GEMM (bf16 WMMA, 64-row tiles): Z, R, preHx, hr
    const int mblocks = (N_NODES + MTILE - 1) / MTILE;
    gemm_gates_kernel<<<mblocks, 256, 0, stream>>>(
        x, h, Lx, Lh, packed, bias, Zbuf, preHx, hrbuf);

    // 7) zero Lhr (aliases Lx), 8) Lhr = L_hat (h*R)
    hipMemsetAsync(Lhr, 0, NF * sizeof(float), stream);
    scatter1_kernel<<<(E_EDGES * 32) / 256, 256, 0, stream>>>(ei, wnorm, hrbuf, Lhr);

    // 9) candidate GEMM + GRU combine + encoder head + pooled partial sums
    gemm_out_kernel<<<mblocks, 256, 0, stream>>>(
        h, hrbuf, Lhr, Zbuf, preHx, packed, bias, out_h0, colsum);

    // 10) out = sigmoid(mean(z))
    finalize_kernel<<<1, 64, 0, stream>>>(colsum, out_head);
}
